// SRNN_34677565948679
// MI455X (gfx1250) — matrix operand order repaired and checked by reading, not compiled, not run on hardware
//
#include <hip/hip_runtime.h>
#include <hip/hip_bf16.h>

#define E_ 62
#define C_ 128
#define H_ 64
#define P_ 16
#define D_ 4
#define B_ 4096

typedef __attribute__((ext_vector_type(16))) __bf16 v16bf;
typedef __attribute__((ext_vector_type(8)))  float  v8f;

#if defined(__has_builtin)
#  if __has_builtin(__builtin_amdgcn_global_load_async_to_lds_b128)
#    define HAVE_ASYNC_LDS 1
#  endif
#  if __has_builtin(__builtin_amdgcn_s_wait_asynccnt)
#    define HAVE_WAIT_ASYNC 1
#  endif
#endif

#if HAVE_ASYNC_LDS
// Builtin signature (from hipcc diagnostic): param0 is AS(1) int-vec4*, so use
// matching pointer types for both the global source and LDS destination.
typedef int async_v4i __attribute__((vector_size(16)));
typedef __attribute__((address_space(1))) async_v4i* gptr128_t;
typedef __attribute__((address_space(3))) async_v4i* sptr128_t;
#endif

// ---------------------------------------------------------------------------
// WMMA fragment helpers (wave32, CDNA5 layouts per CDNA5 ISA)
// ---------------------------------------------------------------------------

// A-matrix 16x32 bf16, row-major source with leading dim ldk (elements).
// lanes 0-15: M=lane, K = {0..7, 16..23}; lanes 16-31: M=lane-16, K = {8..15, 24..31}
__device__ __forceinline__ v16bf load_a_frag(const __bf16* base, int ldk) {
    const int l   = threadIdx.x & 31;
    const int row = l & 15;
    const int kb  = (l >> 4) * 8;
    const __bf16* p = base + (size_t)row * ldk + kb;
    v16bf a;
#pragma unroll
    for (int i = 0; i < 8; ++i) a[i] = p[i];
#pragma unroll
    for (int i = 0; i < 8; ++i) a[8 + i] = p[16 + i];
    return a;
}

// B-matrix 32x16 bf16 where B[k][n] = W[n*ld + k]  (transposed load of row-major W).
// lane = column n (l&15); lanes 0-15 hold K = {0..7, 16..23}, lanes 16-31 hold K = {8..15, 24..31}
__device__ __forceinline__ v16bf load_bT_frag(const __bf16* W, int ld) {
    const int l  = threadIdx.x & 31;
    const int n  = l & 15;
    const int kb = (l >> 4) * 8;
    const __bf16* p = W + (size_t)n * ld + kb;
    v16bf b;
#pragma unroll
    for (int i = 0; i < 8; ++i) { b[i] = p[i]; b[8 + i] = p[16 + i]; }
    return b;
}

// C/D f32 16x16: lane n = l&15; VGPR r holds M = r + 8*(l>=16)
__device__ __forceinline__ v8f load_c_from_bf16(const __bf16* base, int ld, float add) {
    const int l  = threadIdx.x & 31;
    const int n  = l & 15;
    const int mo = (l >> 4) * 8;
    v8f c;
#pragma unroll
    for (int r = 0; r < 8; ++r) c[r] = (float)base[(size_t)(mo + r) * ld + n] + add;
    return c;
}

__device__ __forceinline__ void store_d_bias_bf16(__bf16* base, int ld, v8f d, float add) {
    const int l  = threadIdx.x & 31;
    const int n  = l & 15;
    const int mo = (l >> 4) * 8;
#pragma unroll
    for (int r = 0; r < 8; ++r) base[(size_t)(mo + r) * ld + n] = (__bf16)(d[r] + add);
}

__device__ __forceinline__ void store_d_relu_bf16(__bf16* base, int ld, v8f d) {
    const int l  = threadIdx.x & 31;
    const int n  = l & 15;
    const int mo = (l >> 4) * 8;
#pragma unroll
    for (int r = 0; r < 8; ++r) {
        float v = d[r];
        v = v > 0.f ? v : 0.f;
        base[(size_t)(mo + r) * ld + n] = (__bf16)v;
    }
}

// ---------------------------------------------------------------------------
// Cooperative weight staging: global -> LDS.
// Uses gfx1250 async-to-LDS DMA when the builtin exists (ASYNCcnt tracked),
// else a plain register round-trip. Followed by a workgroup barrier.
// ---------------------------------------------------------------------------
__device__ __forceinline__ void stage_weights(const __bf16* __restrict__ g,
                                              __bf16* s, int bytes) {
    const int nchunk = bytes >> 4;   // 16B per lane per issue
    for (int c = threadIdx.x; c < nchunk; c += (int)blockDim.x) {
        const char* gp = (const char*)g + (size_t)c * 16;
        char* sp = (char*)s + (size_t)c * 16;
#if HAVE_ASYNC_LDS
        __builtin_amdgcn_global_load_async_to_lds_b128(
            (gptr128_t)const_cast<char*>(gp), (sptr128_t)sp, 0, 0);
#else
        *(uint4*)sp = *(const uint4*)gp;
#endif
    }
#if HAVE_ASYNC_LDS
#  if HAVE_WAIT_ASYNC
    __builtin_amdgcn_s_wait_asynccnt(0);
#  else
    asm volatile("s_wait_asynccnt 0x0" ::: "memory");
#  endif
#endif
    __syncthreads();
}

// ---------------------------------------------------------------------------
// Phase 0: precision conversion + graph/weight preprocessing
// ---------------------------------------------------------------------------

__global__ void k_cvt_x(const float* __restrict__ x, __bf16* __restrict__ xb, int n4) {
    int i = blockIdx.x * blockDim.x + threadIdx.x;
    if (i >= n4) return;
    const float4 v = ((const float4*)x)[i];
    __bf16* o = xb + (size_t)i * 4;
    o[0] = (__bf16)v.x; o[1] = (__bf16)v.y; o[2] = (__bf16)v.z; o[3] = (__bf16)v.w;
}

__global__ void k_cvt_w(const float* __restrict__ um, const float* __restrict__ wm,
                        __bf16* __restrict__ uwb, __bf16* __restrict__ wwb, int n) {
    int i = blockIdx.x * blockDim.x + threadIdx.x;
    if (i >= n) return;
    if (i < D_ * H_ * C_) uwb[i] = (__bf16)um[i];
    else                  wwb[i - D_ * H_ * C_] = (__bf16)wm[i - D_ * H_ * C_];
}

// Neighbor lists (<=3, all < step index) from Ns, and collapse the projection:
// w[d,e] = sum_p proj_w[d,p,e]*ps_w[d*P+p];  cbias = sum proj_b*ps_w + ps_b
__global__ void k_prep(const float* __restrict__ Ns, const float* __restrict__ proj_w,
                       const float* __restrict__ proj_b, const float* __restrict__ ps_w,
                       const float* __restrict__ ps_b,
                       int* __restrict__ nbr, float* __restrict__ wproj,
                       float* __restrict__ cbias) {
    const int t = threadIdx.x;
    if (t < D_ * E_) {
        const float* row = Ns + (size_t)t * E_;
        int out[3] = {-1, -1, -1};
        int cnt = 0;
        for (int e = 0; e < E_; ++e)
            if (row[e] != 0.0f && cnt < 3) out[cnt++] = e;
        nbr[t * 3 + 0] = out[0];
        nbr[t * 3 + 1] = out[1];
        nbr[t * 3 + 2] = out[2];

        const int d = t / E_, e = t % E_;
        float acc = 0.f;
        for (int p = 0; p < P_; ++p)
            acc += proj_w[(size_t)(d * P_ + p) * E_ + e] * ps_w[d * P_ + p];
        wproj[t] = acc;
    }
    if (t == 0) {
        float acc = ps_b[0];
        for (int k = 0; k < D_ * P_; ++k) acc += proj_b[k] * ps_w[k];
        *cbias = acc;
    }
}

// ---------------------------------------------------------------------------
// Phase 1: U-GEMM.  u[d,e,b,h] = xb[b, idx[d,e], :] . Uw[d,h,:] + Um_b[d,h]
// Grid (B/128, 8, D). Per block: async-stage Uw[d] (16KB) into LDS once, pull
// all 16 B-frags into VGPRs once, then iterate e (strided by 8): per e only the
// A-fragments stream from global -> 16 back-to-back WMMAs per wave.
// ---------------------------------------------------------------------------

__global__ void __launch_bounds__(256, 1)
k_ugemm(const __bf16* __restrict__ xb, const __bf16* __restrict__ uwb,
        const float* __restrict__ Um_b, const int* __restrict__ dir_idx,
        __bf16* __restrict__ u) {
    __shared__ __bf16 sW[H_ * C_];                 // 16 KB
    const int d = blockIdx.z;
    stage_weights(uwb + (size_t)d * H_ * C_, sW, H_ * C_ * 2);

    const int wave = threadIdx.x >> 5;
    const int b0   = blockIdx.x * 128 + wave * 16;
    const int l    = threadIdx.x & 31;
    const int n    = l & 15;

    v16bf wb[4][4];                                // [ntile][kstep], register-resident
#pragma unroll
    for (int nt = 0; nt < 4; ++nt)
#pragma unroll
        for (int k = 0; k < 4; ++k)
            wb[nt][k] = load_bT_frag(sW + (size_t)(nt * 16) * C_ + k * 32, C_);

    float bias[4];
#pragma unroll
    for (int nt = 0; nt < 4; ++nt) bias[nt] = Um_b[d * H_ + nt * 16 + n];

    for (int e = blockIdx.y; e < E_; e += (int)gridDim.y) {
        const int pt = dir_idx[d * E_ + e];
        const __bf16* Arow = xb + ((size_t)b0 * E_ + pt) * C_;

        const int en = e + (int)gridDim.y;
        if (en < E_) {
            const int ptn = dir_idx[d * E_ + en];
            __builtin_prefetch(xb + ((size_t)b0 * E_ + ptn) * C_, 0, 0);
        }

        v16bf a[4];
#pragma unroll
        for (int k = 0; k < 4; ++k) a[k] = load_a_frag(Arow + k * 32, E_ * C_);

        __bf16* uout = u + (((size_t)d * E_ + e) * B_ + b0) * H_;
#pragma unroll
        for (int nt = 0; nt < 4; ++nt) {
            v8f acc = {};
#pragma unroll
            for (int k = 0; k < 4; ++k)
                acc = __builtin_amdgcn_wmma_f32_16x16x32_bf16(
                    false, a[k], false, wb[nt][k], (short)0, acc, false, false);
            store_d_bias_bf16(uout + nt * 16, H_, acc, bias[nt]);
        }
    }
}

// ---------------------------------------------------------------------------
// Phase 2: recurrence. Each wave owns 16 batch rows for one direction and runs
// all 62 steps: h = relu(u_e + (sum of <=3 prior h) @ Ww^T + Wm_b + sp_bs).
// Ww staged async into LDS, B-frags register-resident across all steps.
// Neighbor aggregation is elementwise per batch row -> all data dependencies
// stay within the wave; same-wave store->load ordering is architectural
// (belt & braces: s_wait_storecnt 0 before re-reading hs).
// ---------------------------------------------------------------------------

__global__ void __launch_bounds__(256, 1)
k_recur(const __bf16* __restrict__ u, const __bf16* __restrict__ wwb,
        const float* __restrict__ Wm_b, const float* __restrict__ sp_bs,
        const int* __restrict__ nbr, __bf16* __restrict__ hs) {
    __shared__ __bf16 sW[H_ * H_];                 // 8 KB
    const int d = blockIdx.y;
    stage_weights(wwb + (size_t)d * H_ * H_, sW, H_ * H_ * 2);

    const int wave = threadIdx.x >> 5;
    const int b0   = blockIdx.x * 128 + wave * 16;
    const int l    = threadIdx.x & 31;
    const int n    = l & 15;

    v16bf wb[4][2];                                // Ww^T fragments, resident 62 steps
#pragma unroll
    for (int nt = 0; nt < 4; ++nt)
#pragma unroll
        for (int k = 0; k < 2; ++k)
            wb[nt][k] = load_bT_frag(sW + (size_t)(nt * 16) * H_ + k * 32, H_);

    float biasn[4];
#pragma unroll
    for (int nt = 0; nt < 4; ++nt)
        biasn[nt] = Wm_b[d * H_ + nt * 16 + n] + sp_bs[d * H_ + nt * 16 + n];

    for (int e = 0; e < E_; ++e) {
        const __bf16* ue = u + (((size_t)d * E_ + e) * B_ + b0) * H_;
        v8f acc[4];
#pragma unroll
        for (int nt = 0; nt < 4; ++nt)
            acc[nt] = load_c_from_bf16(ue + nt * 16, H_, biasn[nt]);

        const int nb0 = nbr[(d * E_ + e) * 3 + 0];
        const int nb1 = nbr[(d * E_ + e) * 3 + 1];
        const int nb2 = nbr[(d * E_ + e) * 3 + 2];
        const int nbrs[3] = {nb0, nb1, nb2};

        // Make prior-step hs stores visible before re-reading (same wave).
        asm volatile("s_wait_storecnt 0x0" ::: "memory");

        float a0[16], a1[16];
#pragma unroll
        for (int i = 0; i < 16; ++i) { a0[i] = 0.f; a1[i] = 0.f; }
#pragma unroll
        for (int j = 0; j < 3; ++j) {
            const int ni = nbrs[j];                // uniform across the wave
            if (ni >= 0) {
                const __bf16* hb = hs + (((size_t)d * E_ + ni) * B_ + b0) * H_;
                v16bf t0 = load_a_frag(hb, H_);
                v16bf t1 = load_a_frag(hb + 32, H_);
#pragma unroll
                for (int i = 0; i < 16; ++i) { a0[i] += (float)t0[i]; a1[i] += (float)t1[i]; }
            }
        }
        v16bf af0, af1;
#pragma unroll
        for (int i = 0; i < 16; ++i) { af0[i] = (__bf16)a0[i]; af1[i] = (__bf16)a1[i]; }

#pragma unroll
        for (int nt = 0; nt < 4; ++nt) {
            acc[nt] = __builtin_amdgcn_wmma_f32_16x16x32_bf16(
                false, af0, false, wb[nt][0], (short)0, acc[nt], false, false);
            acc[nt] = __builtin_amdgcn_wmma_f32_16x16x32_bf16(
                false, af1, false, wb[nt][1], (short)0, acc[nt], false, false);
        }

        __bf16* ho = hs + (((size_t)d * E_ + e) * B_ + b0) * H_;
#pragma unroll
        for (int nt = 0; nt < 4; ++nt)
            store_d_relu_bf16(ho + nt * 16, H_, acc[nt]);
    }
}

// ---------------------------------------------------------------------------
// Phase 3: collapsed projection. out[b,h] = cbias + sum_{d,e} wproj[d,e]*hs[d,e,b,h]
// uint4 (8 x bf16) streaming loads, shift/mask bf16->f32 unpack (no cvt ops);
// pure 130 MB HBM-bound reduction.
// ---------------------------------------------------------------------------

__global__ void k_out(const __bf16* __restrict__ hs, const float* __restrict__ wproj,
                      const float* __restrict__ cbias, float* __restrict__ out) {
    const int nvec = (B_ * H_) / 8;
    const int i = blockIdx.x * blockDim.x + threadIdx.x;
    if (i >= nvec) return;
    const size_t base  = (size_t)i * 8;
    const size_t plane = (size_t)B_ * H_;
    const float cb = *cbias;

    float acc[8];
#pragma unroll
    for (int j = 0; j < 8; ++j) acc[j] = 0.f;

    for (int t = 0; t < D_ * E_; ++t) {
        if (t + 1 < D_ * E_)
            __builtin_prefetch(hs + (size_t)(t + 1) * plane + base, 0, 0);
        const uint4 q = *(const uint4*)(hs + (size_t)t * plane + base);
        const float wt = wproj[t];
        acc[0] += wt * __uint_as_float(q.x << 16);
        acc[1] += wt * __uint_as_float(q.x & 0xffff0000u);
        acc[2] += wt * __uint_as_float(q.y << 16);
        acc[3] += wt * __uint_as_float(q.y & 0xffff0000u);
        acc[4] += wt * __uint_as_float(q.z << 16);
        acc[5] += wt * __uint_as_float(q.z & 0xffff0000u);
        acc[6] += wt * __uint_as_float(q.w << 16);
        acc[7] += wt * __uint_as_float(q.w & 0xffff0000u);
    }
    float4* o = (float4*)(out + base);
    o[0] = make_float4(acc[0] + cb, acc[1] + cb, acc[2] + cb, acc[3] + cb);
    o[1] = make_float4(acc[4] + cb, acc[5] + cb, acc[6] + cb, acc[7] + cb);
}

// ---------------------------------------------------------------------------
// Host launcher
// ---------------------------------------------------------------------------

extern "C" void kernel_launch(void* const* d_in, const int* in_sizes, int n_in,
                              void* d_out, int out_size, void* d_ws, size_t ws_size,
                              hipStream_t stream) {
    (void)in_sizes; (void)n_in; (void)out_size; (void)ws_size;

    const float* x       = (const float*)d_in[0];
    const float* Ns      = (const float*)d_in[1];
    const int*   dir_idx = (const int*)  d_in[2];
    const float* Um_w    = (const float*)d_in[3];
    const float* Um_b    = (const float*)d_in[4];
    const float* Wm_w    = (const float*)d_in[5];
    const float* Wm_b    = (const float*)d_in[6];
    const float* sp_bs   = (const float*)d_in[7];
    const float* proj_w  = (const float*)d_in[8];
    const float* proj_b  = (const float*)d_in[9];
    const float* ps_w    = (const float*)d_in[10];
    const float* ps_b    = (const float*)d_in[11];
    float* out = (float*)d_out;

    size_t off = 0;
    auto carve = [&](size_t bytes) -> void* {
        void* p = (void*)((char*)d_ws + off);
        off += (bytes + 255) & ~(size_t)255;
        return p;
    };
    __bf16* xb    = (__bf16*)carve((size_t)B_ * E_ * C_ * 2);        // x in bf16
    __bf16* uwb   = (__bf16*)carve((size_t)D_ * H_ * C_ * 2);        // Um_w bf16
    __bf16* wwb   = (__bf16*)carve((size_t)D_ * H_ * H_ * 2);        // Wm_w bf16
    __bf16* ubuf  = (__bf16*)carve((size_t)D_ * E_ * B_ * H_ * 2);   // u (incl Um_b)
    __bf16* hs    = (__bf16*)carve((size_t)D_ * E_ * B_ * H_ * 2);   // hidden history
    int*    nbr   = (int*)   carve((size_t)D_ * E_ * 3 * 4);         // neighbor lists
    float*  wproj = (float*) carve((size_t)(D_ * E_ + 1) * 4);       // collapsed proj
    float*  cbias = wproj + D_ * E_;

    {   // x fp32 -> bf16 (float4-wide)
        int n4 = (B_ * E_ * C_) / 4;
        k_cvt_x<<<(n4 + 255) / 256, 256, 0, stream>>>(x, xb, n4);
    }
    {   // weights fp32 -> bf16
        int n = D_ * H_ * C_ + D_ * H_ * H_;
        k_cvt_w<<<(n + 255) / 256, 256, 0, stream>>>(Um_w, Wm_w, uwb, wwb, n);
    }
    k_prep<<<1, 256, 0, stream>>>(Ns, proj_w, proj_b, ps_w, ps_b, nbr, wproj, cbias);

    {   // U-GEMM: 16.6 GFLOP through v_wmma_f32_16x16x32_bf16
        dim3 g(B_ / 128, 8, D_);
        k_ugemm<<<g, 256, 0, stream>>>(xb, uwb, Um_b, dir_idx, ubuf);
    }
    {   // 62-step recurrence, 8 WMMA per wave per step
        dim3 g(B_ / 128, D_);
        k_recur<<<g, 256, 0, stream>>>(ubuf, wwb, Wm_b, sp_bs, nbr, hs);
    }
    {   // collapsed projection + final dot (HBM-bound streaming reduction)
        int n = (B_ * H_) / 8;
        k_out<<<(n + 255) / 256, 256, 0, stream>>>(hs, wproj, cbias, out);
    }
}
